// SequenceEncoder_28054726378041
// MI455X (gfx1250) — matrix operand order repaired and checked by reading, not compiled, not run on hardware
//
#include <hip/hip_runtime.h>

// ---------------------------------------------------------------------------
// GRU sequence encoder for MI455X (gfx1250, wave32, WMMA f16 16x16x32).
// One wave owns a 16-row tile; all weights persistent in VGPRs; rows counting-
// sorted by length so each wave only iterates to its tile's max length.
// v3: process jc-groups in pairs -> two independent WMMA chains + elementwise
//     streams in flight (fills WMMA->VALU hazard slots, enables pk/dual-issue)
//     while staying under the 512-VGPR / 2-waves-per-SIMD cap.
// ---------------------------------------------------------------------------

typedef __attribute__((ext_vector_type(16))) _Float16 v16h;
typedef __attribute__((ext_vector_type(2)))  __fp16   v2fp;
typedef __attribute__((ext_vector_type(8)))  float    v8f;

#define NROWS 16384
#define LSEQ  512
#define VOC   32000
#define ES    32
#define HSZ   64

union H16x16 { uint4 q[2]; v16h h; };
union PackH  { _Float16 h[2]; unsigned u; };
union Pack2  { v2fp h2; unsigned u; };

static __device__ __forceinline__ float tanh_fast(float x) {
#if __has_builtin(__builtin_amdgcn_tanhf)
  return __builtin_amdgcn_tanhf(x);            // v_tanh_f32 (CDNA5 trans op)
#else
  float e = __expf(2.0f * x);
#if __has_builtin(__builtin_amdgcn_rcpf)
  return 1.0f - 2.0f * __builtin_amdgcn_rcpf(e + 1.0f);
#else
  return 1.0f - 2.0f / (e + 1.0f);
#endif
#endif
}

static __device__ __forceinline__ unsigned pack_f16x2(float lo, float hi) {
#if __has_builtin(__builtin_amdgcn_cvt_pkrtz)
  Pack2 p; p.h2 = __builtin_amdgcn_cvt_pkrtz(lo, hi); return p.u;
#else
  PackH p; p.h[0] = (_Float16)lo; p.h[1] = (_Float16)hi; return p.u;
#endif
}

// Hidden-dim permutation: WMMA A-operand K index -> actual hidden unit.
// Chosen so that a lane's cvt_pk dword (cols c, c+16) stored at LDS dword
// d = p*16 + c%16 reloads directly as the f16 A-operand K pair.
static __device__ __forceinline__ int psi_k(int k) {
  int o = k >> 5, rem = k & 31, hh = rem >> 4, q = rem & 15, v = q >> 1, b = q & 1;
  return o * 32 + hh * 8 + v + 16 * b;
}

// ---------------------------------------------------------------------------
// Prep 1: per-row length (binary search over the zero-padded suffix) + histogram.
// ---------------------------------------------------------------------------
__global__ void gru_lenhist(const int* __restrict__ x, int* __restrict__ lengths,
                            int* __restrict__ hist) {
  int r = blockIdx.x * blockDim.x + threadIdx.x;
  if (r >= NROWS) return;
  const int* row = x + (size_t)r * LSEQ;
  int lo = 0, hi = LSEQ;
  while (lo < hi) { int mid = (lo + hi) >> 1; if (row[mid] != 0) lo = mid + 1; else hi = mid; }
  lengths[r] = lo;
  atomicAdd(&hist[lo], 1);
}

// Prep 2: descending exclusive scan -> longest rows get lowest offsets.
__global__ void gru_scan(const int* __restrict__ hist, int* __restrict__ off) {
  if (threadIdx.x == 0 && blockIdx.x == 0) {
    int run = 0;
    for (int l = LSEQ; l >= 0; --l) { off[l] = run; run += hist[l]; }
  }
}

// Prep 3: scatter rows into length-sorted order.
__global__ void gru_perm(const int* __restrict__ lengths, int* __restrict__ off,
                         int* __restrict__ perm) {
  int r = blockIdx.x * blockDim.x + threadIdx.x;
  if (r >= NROWS) return;
  int pos = atomicAdd(&off[lengths[r]], 1);
  perm[pos] = r;
}

// ---------------------------------------------------------------------------
// Prep 4: f16 embedding + weight pre-swizzle into this code's contiguous-K WMMA f16 B-operand order
// (K=32,N=16): lanes 0-15: N=lane, VGPR v: K=2v,2v+1; lanes 16-31: N=lane-16,
// VGPR v: K=16+2v,17+2v.  r/z blocks (rows<128) scaled by 0.5 (sigmoid fold).
// w_hh additionally has psi_k applied to its K axis (h A-operand permutation).
// Bias: 16 tiles x 16 cols (r,z: 0.5*(bih+bhh); xn: bih; hn: bhh).
// ---------------------------------------------------------------------------
#define EMB_ELEMS (VOC * ES)
#define WIH_DW    (12 * 32 * 8)
#define WHH_DW    (12 * 2 * 32 * 8)
#define BIAS_EL   (16 * 16)

__global__ void gru_pack(const float* __restrict__ emb, const float* __restrict__ w_ih,
                         const float* __restrict__ w_hh, const float* __restrict__ b_ih,
                         const float* __restrict__ b_hh, _Float16* __restrict__ emb16,
                         unsigned* __restrict__ wihB, unsigned* __restrict__ whhB,
                         float* __restrict__ biasT) {
  int i = blockIdx.x * blockDim.x + threadIdx.x;
  if (i < EMB_ELEMS) { emb16[i] = (_Float16)emb[i]; return; }
  i -= EMB_ELEMS;
  if (i < WIH_DW) {
    int j = i >> 8, rem = i & 255, lane = rem >> 3, v = rem & 7;
    int n = 16 * j + (lane & 15);
    int k0 = (lane < 16) ? (2 * v) : (16 + 2 * v);
    float s = (n < 128) ? 0.5f : 1.0f;   // r/z gates pre-scaled for sigmoid
    PackH p;
    p.h[0] = (_Float16)(s * w_ih[n * ES + k0]);
    p.h[1] = (_Float16)(s * w_ih[n * ES + k0 + 1]);
    wihB[i] = p.u;
    return;
  }
  i -= WIH_DW;
  if (i < WHH_DW) {
    int j = i >> 9, rem = i & 511, c = rem >> 8, rem2 = rem & 255;
    int lane = rem2 >> 3, v = rem2 & 7;
    int n = 16 * j + (lane & 15);
    int k0 = 32 * c + ((lane < 16) ? (2 * v) : (16 + 2 * v));
    float s = (n < 128) ? 0.5f : 1.0f;
    PackH p;
    p.h[0] = (_Float16)(s * w_hh[n * HSZ + psi_k(k0)]);
    p.h[1] = (_Float16)(s * w_hh[n * HSZ + psi_k(k0 + 1)]);
    whhB[i] = p.u;
    return;
  }
  i -= WHH_DW;
  if (i < BIAS_EL) {
    int tile = i >> 4, col = i & 15;
    float val;
    if (tile < 4)       { int row = tile * 16 + col;              val = 0.5f * (b_ih[row] + b_hh[row]); }
    else if (tile < 8)  { int row = 64 + (tile - 4) * 16 + col;   val = 0.5f * (b_ih[row] + b_hh[row]); }
    else if (tile < 12) { int row = 128 + (tile - 8) * 16 + col;  val = b_ih[row]; }
    else                { int row = 128 + (tile - 12) * 16 + col; val = b_hh[row]; }
    biasT[i] = val;
  }
}

// ---------------------------------------------------------------------------
// Main kernel: 1024 single-wave workgroups, each runs a 16-row GRU tile.
// ---------------------------------------------------------------------------
__global__ __launch_bounds__(32, 2) void gru_main(
    const int* __restrict__ x, const _Float16* __restrict__ emb16,
    const uint4* __restrict__ wihB, const uint4* __restrict__ whhB,
    const float* __restrict__ biasT, const int* __restrict__ perm,
    const int* __restrict__ lens, float* __restrict__ out) {
  const int lane = threadIdx.x;
  const int tile = blockIdx.x;
  const int col  = lane & 15;
  const int hi   = (lane >> 4) & 1;

  // h round-trip buffer: 16 rows x 32 dwords (64 f16), row stride 36 dwords.
  __shared__ alignas(16) unsigned hd[16 * 36];

  // ---- persistent weights in VGPRs -----------------------------------------
  v16h wih[12];
#pragma unroll
  for (int j = 0; j < 12; ++j) {
    H16x16 u;
    const uint4* p = wihB + (j * 32 + lane) * 2;
    u.q[0] = p[0]; u.q[1] = p[1];
    wih[j] = u.h;
  }
  v16h whh[24];  // [tile jj=0..11][kchunk c=0..1] flattened jj*2+c
#pragma unroll
  for (int j = 0; j < 24; ++j) {
    H16x16 u;
    const uint4* p = whhB + (j * 32 + lane) * 2;
    u.q[0] = p[0]; u.q[1] = p[1];
    whh[j] = u.h;
  }
  float bR[4], bZ[4], bXN[4], bHN[4];
#pragma unroll
  for (int jc = 0; jc < 4; ++jc) {
    bR[jc]  = biasT[(jc)      * 16 + col];
    bZ[jc]  = biasT[(jc + 4)  * 16 + col];
    bXN[jc] = biasT[(jc + 8)  * 16 + col];
    bHN[jc] = biasT[(jc + 12) * 16 + col];
  }

  // ---- row bookkeeping (length-sorted tiles) -------------------------------
  const int base = tile * 16;
  const int rowTok = perm[base + col];            // row for A-operand lane (M = lane&15)
  int rowC[8], lenC[8];                           // C-layout rows: M = v + hi*8
#pragma unroll
  for (int v = 0; v < 8; ++v) {
    int rg = perm[base + v + hi * 8];
    rowC[v] = rg;
    lenC[v] = lens[rg];
  }
  const int tmax = __builtin_amdgcn_readfirstlane(lens[rowTok]);        // longest row
  const int tmin = __builtin_amdgcn_readfirstlane(lens[perm[base + 15]]); // shortest row

  // ---- h state -------------------------------------------------------------
  float h[4][8];
#pragma unroll
  for (int jc = 0; jc < 4; ++jc)
#pragma unroll
    for (int v = 0; v < 8; ++v) h[jc][v] = 0.0f;
  for (int i = lane; i < 16 * 36; i += 32) hd[i] = 0u;
  __syncthreads();

  // A-operand reload: operand o at byte o*64; lane-half at +hi*32; v0-3/+16 v4-7
  const char* hrow = (const char*)hd + col * 144 + hi * 32;
  auto loadHA = [&](v16h& a0, v16h& a1) {
    H16x16 u0, u1;
    u0.q[0] = *(const uint4*)(hrow + 0);   u0.q[1] = *(const uint4*)(hrow + 16);
    u1.q[0] = *(const uint4*)(hrow + 64);  u1.q[1] = *(const uint4*)(hrow + 80);
    a0 = u0.h; a1 = u1.h;
  };
  auto loadE = [&](int tok, v16h& e) {
    const char* ep = (const char*)emb16 + (size_t)tok * 64 + hi * 32;
    H16x16 ue;
    ue.q[0] = *(const uint4*)ep;
    ue.q[1] = *(const uint4*)(ep + 16);
    e = ue.h;
  };
  auto mma = [&](v16h a, v16h b, v8f c) -> v8f {
    return __builtin_amdgcn_wmma_f32_16x16x32_f16(false, a, false, b, (short)0, c,
                                                  false, false);
  };

  v16h hA0, hA1;
  loadHA(hA0, hA1);

  const int* xrow = x + (size_t)rowTok * LSEQ;

  auto step = [&](int t, v16h e, bool masked) {
    v8f zero = {};
    // jc-pair fusion: two independent 9-WMMA chains + two elementwise streams
    // in flight -> scheduler can fill WMMA->VALU hazard slots with real work.
#pragma unroll
    for (int jp = 0; jp < 2; ++jp) {
      const int jA = 2 * jp, jB = 2 * jp + 1;
      v8f aRA  = mma(e, wih[jA],     mma(hA1, whh[(jA)     * 2 + 1], mma(hA0, whh[(jA)     * 2], zero)));
      v8f aRB  = mma(e, wih[jB],     mma(hA1, whh[(jB)     * 2 + 1], mma(hA0, whh[(jB)     * 2], zero)));
      v8f aZA  = mma(e, wih[jA + 4], mma(hA1, whh[(jA + 4) * 2 + 1], mma(hA0, whh[(jA + 4) * 2], zero)));
      v8f aZB  = mma(e, wih[jB + 4], mma(hA1, whh[(jB + 4) * 2 + 1], mma(hA0, whh[(jB + 4) * 2], zero)));
      v8f aHNA =                     mma(hA1, whh[(jA + 8) * 2 + 1], mma(hA0, whh[(jA + 8) * 2], zero));
      v8f aHNB =                     mma(hA1, whh[(jB + 8) * 2 + 1], mma(hA0, whh[(jB + 8) * 2], zero));
      v8f aXNA = mma(e, wih[jA + 8], zero);
      v8f aXNB = mma(e, wih[jB + 8], zero);
#pragma unroll
      for (int v = 0; v < 8; ++v) {
        float rA  = __builtin_fmaf(tanh_fast(aRA[v] + bR[jA]), 0.5f, 0.5f);
        float rB  = __builtin_fmaf(tanh_fast(aRB[v] + bR[jB]), 0.5f, 0.5f);
        float zA  = __builtin_fmaf(tanh_fast(aZA[v] + bZ[jA]), 0.5f, 0.5f);
        float zB  = __builtin_fmaf(tanh_fast(aZB[v] + bZ[jB]), 0.5f, 0.5f);
        float ngA = tanh_fast(__builtin_fmaf(rA, aHNA[v] + bHN[jA], aXNA[v] + bXN[jA]));
        float ngB = tanh_fast(__builtin_fmaf(rB, aHNB[v] + bHN[jB], aXNB[v] + bXN[jB]));
        float hvA = h[jA][v], hvB = h[jB][v];
        float hnA = __builtin_fmaf(zA, hvA - ngA, ngA);
        float hnB = __builtin_fmaf(zB, hvB - ngB, ngB);
        if (masked) {
          hnA = (t < lenC[v]) ? hnA : hvA;
          hnB = (t < lenC[v]) ? hnB : hvB;
        }
        h[jA][v] = hnA;
        h[jB][v] = hnB;
        // pack (col, col+16) pair for this jc-pair straight into LDS dword
        hd[(v + hi * 8) * 36 + jp * 16 + col] = pack_f16x2(hnA, hnB);
      }
    }
    __syncthreads();
    loadHA(hA0, hA1);
  };

  int2 tk = *(const int2*)(xrow);
  v16h e0, e1;
  loadE(tk.x, e0);
  loadE(tk.y, e1);

  int t0 = 0;
  // fast loop: every row in the tile is still live -> no masking
  for (; t0 + 1 < tmin; t0 += 2) {
    int tnext = (t0 + 2 < LSEQ) ? (t0 + 2) : (LSEQ - 2);
    int2 tkn = *(const int2*)(xrow + tnext);
    step(t0, e0, false);
    loadE(tkn.x, e0);           // prefetch a full step ahead
    step(t0 + 1, e1, false);
    loadE(tkn.y, e1);
  }
  // tail: per-row freeze masking
  for (; t0 < tmax; t0 += 2) {
    int tnext = (t0 + 2 < LSEQ) ? (t0 + 2) : (LSEQ - 2);
    int2 tkn = *(const int2*)(xrow + tnext);
    step(t0, e0, true);
    loadE(tkn.x, e0);
    step(t0 + 1, e1, true);
    loadE(tkn.y, e1);
  }

  // scatter final hidden states back to original row order
#pragma unroll
  for (int jc = 0; jc < 4; ++jc)
#pragma unroll
    for (int v = 0; v < 8; ++v)
      out[(size_t)rowC[v] * HSZ + jc * 16 + col] = h[jc][v];
}

// ---------------------------------------------------------------------------
// Launch: inputs = { x, embedding, w_ih, w_hh, b_ih, b_hh }
// ---------------------------------------------------------------------------
extern "C" void kernel_launch(void* const* d_in, const int* in_sizes, int n_in,
                              void* d_out, int out_size, void* d_ws, size_t ws_size,
                              hipStream_t stream) {
  const int*   x    = (const int*)d_in[0];
  const float* emb  = (const float*)d_in[1];
  const float* w_ih = (const float*)d_in[2];
  const float* w_hh = (const float*)d_in[3];
  const float* b_ih = (const float*)d_in[4];
  const float* b_hh = (const float*)d_in[5];
  float* out = (float*)d_out;

  char* ws = (char*)d_ws;
  size_t o = 0;
  _Float16* emb16  = (_Float16*)(ws + o); o += (size_t)EMB_ELEMS * 2;   // 2,048,000
  unsigned* wihB   = (unsigned*)(ws + o); o += (size_t)WIH_DW * 4;      // 12,288
  unsigned* whhB   = (unsigned*)(ws + o); o += (size_t)WHH_DW * 4;      // 24,576
  float*    biasT  = (float*)(ws + o);    o += (size_t)BIAS_EL * 4;     // 1,024
  int*      lengths= (int*)(ws + o);      o += (size_t)NROWS * 4;       // 65,536
  int*      hist   = (int*)(ws + o);      o += 2304;
  int*      off    = (int*)(ws + o);      o += 2304;
  int*      perm   = (int*)(ws + o);      o += (size_t)NROWS * 4;

  (void)hipMemsetAsync(hist, 0, 2304, stream);
  gru_lenhist<<<NROWS / 256, 256, 0, stream>>>(x, lengths, hist);
  gru_scan<<<1, 32, 0, stream>>>(hist, off);
  gru_perm<<<NROWS / 256, 256, 0, stream>>>(lengths, off, perm);

  const int packTotal = EMB_ELEMS + WIH_DW + WHH_DW + BIAS_EL;
  gru_pack<<<(packTotal + 255) / 256, 256, 0, stream>>>(emb, w_ih, w_hh, b_ih, b_hh,
                                                        emb16, wihB, whhB, biasT);

  gru_main<<<NROWS / 16, 32, 0, stream>>>(x, emb16, (const uint4*)wihB,
                                          (const uint4*)whhB, biasT, perm, lengths, out);
}
